// RoomCritic_88673894793689
// MI455X (gfx1250) — matrix operand order repaired and checked by reading, compiled, not run on hardware
//
#include <hip/hip_runtime.h>
#include <hip/hip_bf16.h>

// ---------- types / helpers ----------
typedef __attribute__((ext_vector_type(16))) __bf16 v16bf;
typedef __attribute__((ext_vector_type(8)))  float  v8f;

union Frag { v16bf v; uint4 q[2]; };

__device__ __forceinline__ unsigned short f2bf(float f) {
  unsigned u = __float_as_uint(f);
  u += 0x7FFFu + ((u >> 16) & 1u);   // round-to-nearest-even
  return (unsigned short)(u >> 16);
}
__device__ __forceinline__ float bf2f(unsigned short h) {
  return __uint_as_float(((unsigned)h) << 16);
}
// order-preserving float->uint key for atomic max
__device__ __forceinline__ unsigned enc_key(float f) {
  unsigned b = __float_as_uint(f);
  return (b & 0x80000000u) ? ~b : (b | 0x80000000u);
}
__device__ __forceinline__ float dec_key(unsigned u) {
  if (u == 0u) return 0.0f;                       // never-written sentinel
  unsigned bits = (u & 0x80000000u) ? (u & 0x7FFFFFFFu) : ~u;
  float f = __uint_as_float(bits);
  if (!(f > -3.0e38f && f < 3.0e38f)) return 0.0f; // non-finite -> 0 (matches reference)
  return f;
}

#define ET  16          // edges per workgroup (one 16-row M tile)
#define AST 912         // A_lds row stride (896 + 16 pad), ushorts
#define HST 272         // hidden row stride (256 + 16 pad), ushorts

// ---------- small node-level kernels (f32 VALU, negligible FLOPs) ----------

// wall MLP 1->64->64, outer tanh. one block per batch element, 64 threads.
__global__ void rc_wall(const float* __restrict__ wall, const float* __restrict__ w1,
                        const float* __restrict__ b1, const float* __restrict__ w2,
                        const float* __restrict__ b2, float* __restrict__ out) {
  __shared__ float h[64];
  int b = blockIdx.x, t = threadIdx.x;
  float w = wall[b];
  h[t] = tanhf(w * w1[t] + b1[t]);
  __syncthreads();
  float o = b2[t];
  for (int k = 0; k < 64; ++k) o += h[k] * w2[k * 64 + t];
  out[b * 64 + t] = tanhf(o);
}

// per-node cond = [class(64) | wall(64) | geo(64)]; write f32 cond + bf16 into hnode cols 256..447
__global__ void rc_cond(const int* __restrict__ category, const float* __restrict__ geo,
                        const int* __restrict__ batch, const float* __restrict__ cat_emb,
                        const float* __restrict__ cw, const float* __restrict__ cb,
                        const float* __restrict__ gw1, const float* __restrict__ gb1,
                        const float* __restrict__ gw2, const float* __restrict__ gb2,
                        const float* __restrict__ wallf, float* __restrict__ cond,
                        unsigned short* __restrict__ hnode) {
  __shared__ float tcls[64], hg[64];
  int n = blockIdx.x, t = threadIdx.x;
  int cat = category[n];
  tcls[t] = tanhf(tanhf(cat_emb[cat * 64 + t]));
  float g0 = geo[n * 2 + 0], g1 = geo[n * 2 + 1];
  hg[t] = tanhf(g0 * gw1[t] + g1 * gw1[64 + t] + gb1[t]);
  __syncthreads();
  float cl = cb[t], ge = gb2[t];
  for (int k = 0; k < 64; ++k) {
    cl += tcls[k] * cw[k * 64 + t];
    ge += hg[k] * gw2[k * 64 + t];
  }
  cl = tanhf(cl); ge = tanhf(ge);
  float wl = wallf[batch[n] * 64 + t];
  float* cn = cond + (size_t)n * 192;
  cn[t] = cl; cn[64 + t] = wl; cn[128 + t] = ge;
  unsigned short* hn = hnode + (size_t)n * 448;
  hn[256 + t] = f2bf(cl); hn[320 + t] = f2bf(wl); hn[384 + t] = f2bf(ge);
}

// init MLP 10->256->256, outer tanh; bf16 into hnode cols 0..255. 256 thr / node.
__global__ void rc_init(const float* __restrict__ x, const float* __restrict__ act,
                        const float* __restrict__ tar, const float* __restrict__ w1,
                        const float* __restrict__ b1, const float* __restrict__ w2,
                        const float* __restrict__ b2, unsigned short* __restrict__ hnode) {
  __shared__ float in10[10];
  __shared__ float h[256];
  int n = blockIdx.x, t = threadIdx.x;
  if (t < 4) in10[t] = x[n * 4 + t];
  else if (t < 7) in10[t] = act[n * 3 + (t - 4)];
  else if (t < 10) in10[t] = tar[n * 3 + (t - 7)];
  __syncthreads();
  float a = b1[t];
  #pragma unroll
  for (int k = 0; k < 10; ++k) a += in10[k] * w1[k * 256 + t];
  h[t] = tanhf(a);
  __syncthreads();
  float o = b2[t];
  for (int k = 0; k < 256; ++k) o += h[k] * w2[k * 256 + t];
  hnode[(size_t)n * 448 + t] = f2bf(tanhf(o));
}

// transpose + f32->bf16: wt[j*K+k] = w[k*256+j]   (w is [K,256])
__global__ void rc_tw(const float* __restrict__ w, unsigned short* __restrict__ wt, int K) {
  int id = blockIdx.x * 256 + threadIdx.x;
  if (id < K * 256) {
    int j = id / K, k = id - j * K;
    wt[id] = f2bf(w[k * 256 + j]);
  }
}

__global__ void rc_zero(unsigned* __restrict__ p, int n) {
  int i = blockIdx.x * 256 + threadIdx.x;
  if (i < n) p[i] = 0u;
}

// decode conv1 segment-max, tanh, overwrite hnode cols 0..255 (cond cols unchanged)
__global__ void rc_post(const unsigned* __restrict__ out_enc, unsigned short* __restrict__ hnode) {
  int n = blockIdx.x, c = threadIdx.x;
  hnode[(size_t)n * 448 + c] = f2bf(tanhf(dec_key(out_enc[(size_t)n * 256 + c])));
}

// tail MLP 448->256->1 over [tanh(conv2max) | cond]
__global__ void rc_tail(const unsigned* __restrict__ out_enc, const float* __restrict__ cond,
                        const float* __restrict__ w1, const float* __restrict__ b1,
                        const float* __restrict__ w2, const float* __restrict__ b2,
                        float* __restrict__ out) {
  __shared__ float g[448];
  __shared__ float red[256];
  int n = blockIdx.x, t = threadIdx.x;
  g[t] = tanhf(dec_key(out_enc[(size_t)n * 256 + t]));
  if (t < 192) g[256 + t] = cond[(size_t)n * 192 + t];
  __syncthreads();
  float a = b1[t];
  for (int k = 0; k < 448; ++k) a += g[k] * w1[k * 256 + t];
  red[t] = tanhf(a) * w2[t];
  __syncthreads();
  for (int s = 128; s > 0; s >>= 1) {
    if (t < s) red[t] += red[t + s];
    __syncthreads();
  }
  if (t == 0) out[n] = red[0] + b2[0];
}

// ---------- the WMMA EdgeConv: per-edge MLP 896->256->256 + atomic segment-max ----------
__global__ __launch_bounds__(256)
void rc_edgeconv(const unsigned short* __restrict__ hnode, // [N,448] bf16
                 const unsigned short* __restrict__ W1t,   // [256,896] bf16 (col-major of l1.w)
                 const float* __restrict__ b1,
                 const unsigned short* __restrict__ W2t,   // [256,256] bf16
                 const float* __restrict__ b2,
                 const int* __restrict__ src, const int* __restrict__ dst,
                 unsigned* __restrict__ out_enc, int E) {
  __shared__ __align__(16) unsigned short Alds[ET * AST]; // 16 x 896 edge inputs
  __shared__ __align__(16) unsigned short Hlds[ET * HST]; // 16 x 256 hidden
  int tid = threadIdx.x;
  int lane = tid & 31, wave = tid >> 5;
  int tileBase = blockIdx.x * ET;

  // gather [x_i | x_j - x_i] for 16 edges into LDS (4 bf16 per step)
  for (int g = tid; g < ET * 224; g += 256) {
    int e_l = g / 224;
    int k = (g - e_l * 224) * 4;
    int eg = tileBase + e_l; if (eg >= E) eg = E - 1;
    int i = dst[eg], j = src[eg];
    if (k < 448) {
      uint2 v = *(const uint2*)(hnode + (size_t)i * 448 + k);
      *(uint2*)(&Alds[e_l * AST + k]) = v;
    } else {
      int kk = k - 448;
      uint2 vi = *(const uint2*)(hnode + (size_t)i * 448 + kk);
      uint2 vj = *(const uint2*)(hnode + (size_t)j * 448 + kk);
      const unsigned short* pi = (const unsigned short*)&vi;
      const unsigned short* pj = (const unsigned short*)&vj;
      unsigned short r[4];
      #pragma unroll
      for (int q = 0; q < 4; ++q) r[q] = f2bf(bf2f(pj[q]) - bf2f(pi[q]));
      *(uint2*)(&Alds[e_l * AST + k]) = *(uint2*)r;
    }
  }
  __syncthreads();

  // A operand k order used here (contiguous, same as B): lanes 0-15 hold K 0..15, lanes 16-31 hold K 16..31
  int rowA  = lane & 15;
  int halfA = (lane & 16) ? 16 : 0;
  // B operand k order used here (contiguous, same as A): lanes 0-15 = K 0..15, lanes 16-31 = K 16..31 (col = lane&15)
  int halfB = (lane & 16) ? 16 : 0;
  int col0 = wave * 32 + (lane & 15);
  int col1 = col0 + 16;

  // ---- layer 1: [16,896] @ [896, 32 cols per wave] ----
  v8f c0 = {}; v8f c1 = {};
  for (int k0 = 0; k0 < 896; k0 += 32) {
    Frag a, bA, bB;
    a.q[0] = *(const uint4*)(&Alds[rowA * AST + k0 + halfA]);
    a.q[1] = *(const uint4*)(&Alds[rowA * AST + k0 + halfA + 8]);
    const uint4* p0 = (const uint4*)(W1t + (size_t)col0 * 896 + k0 + halfB);
    bA.q[0] = p0[0]; bA.q[1] = p0[1];
    const uint4* p1 = (const uint4*)(W1t + (size_t)col1 * 896 + k0 + halfB);
    bB.q[0] = p1[0]; bB.q[1] = p1[1];
    c0 = __builtin_amdgcn_wmma_f32_16x16x32_bf16(false, a.v, false, bA.v, (short)0, c0, false, false);
    c1 = __builtin_amdgcn_wmma_f32_16x16x32_bf16(false, a.v, false, bB.v, (short)0, c1, false, false);
  }
  // bias + tanh -> hidden LDS (C layout: vgpr r, lanes<16 -> M=r, lanes>=16 -> M=r+8)
  int moff = (lane & 16) ? 8 : 0;
  float bb0 = b1[col0], bb1 = b1[col1];
  #pragma unroll
  for (int r = 0; r < 8; ++r) {
    int m = r + moff;
    Hlds[m * HST + col0] = f2bf(tanhf(c0[r] + bb0));
    Hlds[m * HST + col1] = f2bf(tanhf(c1[r] + bb1));
  }
  __syncthreads();

  // ---- layer 2: [16,256] @ [256, 32 cols per wave] ----
  v8f d0 = {}; v8f d1 = {};
  for (int k0 = 0; k0 < 256; k0 += 32) {
    Frag a, bA, bB;
    a.q[0] = *(const uint4*)(&Hlds[rowA * HST + k0 + halfA]);
    a.q[1] = *(const uint4*)(&Hlds[rowA * HST + k0 + halfA + 8]);
    const uint4* p0 = (const uint4*)(W2t + (size_t)col0 * 256 + k0 + halfB);
    bA.q[0] = p0[0]; bA.q[1] = p0[1];
    const uint4* p1 = (const uint4*)(W2t + (size_t)col1 * 256 + k0 + halfB);
    bB.q[0] = p1[0]; bB.q[1] = p1[1];
    d0 = __builtin_amdgcn_wmma_f32_16x16x32_bf16(false, a.v, false, bA.v, (short)0, d0, false, false);
    d1 = __builtin_amdgcn_wmma_f32_16x16x32_bf16(false, a.v, false, bB.v, (short)0, d1, false, false);
  }
  // bias + atomic segment-max into dst rows
  float cb0 = b2[col0], cb1 = b2[col1];
  #pragma unroll
  for (int r = 0; r < 8; ++r) {
    int m = r + moff;
    int eg = tileBase + m;
    if (eg < E) {
      int dn = dst[eg];
      atomicMax(&out_enc[(size_t)dn * 256 + col0], enc_key(d0[r] + cb0));
      atomicMax(&out_enc[(size_t)dn * 256 + col1], enc_key(d1[r] + cb1));
    }
  }
}

// ---------- launch ----------
extern "C" void kernel_launch(void* const* d_in, const int* in_sizes, int n_in,
                              void* d_out, int out_size, void* d_ws, size_t ws_size,
                              hipStream_t stream) {
  const float* wall_batch = (const float*)d_in[0];
  const float* x          = (const float*)d_in[1];
  const float* actions    = (const float*)d_in[2];
  const float* tars       = (const float*)d_in[3];
  const float* geo        = (const float*)d_in[4];
  const int*   category   = (const int*)d_in[5];
  const int*   batch      = (const int*)d_in[6];
  const int*   eidx       = (const int*)d_in[7];
  int B = in_sizes[0];
  int N = in_sizes[5];
  int E = in_sizes[7] / 2;
  const int* src = eidx;
  const int* dst = eidx + E;

  size_t off = 0;
  auto alloc = [&](size_t bytes) -> void* {
    void* p = (char*)d_ws + off;
    off += (bytes + 255) & ~(size_t)255;
    return p;
  };
  float*          wallf   = (float*)alloc((size_t)B * 64 * 4);
  float*          cond    = (float*)alloc((size_t)N * 192 * 4);
  unsigned short* hnode   = (unsigned short*)alloc((size_t)N * 448 * 2);
  unsigned*       out_enc = (unsigned*)alloc((size_t)N * 256 * 4);
  unsigned short* W1t     = (unsigned short*)alloc((size_t)256 * 896 * 2);
  unsigned short* W2t     = (unsigned short*)alloc((size_t)256 * 256 * 2);

  float* q = (float*)d_out;

  for (int p = 0; p < 2; ++p) {
    // params flattened as jax pytree (sorted dict keys), 27 leaves per set
    int base = 8 + p * 27;
    auto P = [&](int i) { return (const float*)d_in[base + i]; };
    const float* cat_emb = P(0);
    const float* cl_b = P(1);  const float* cl_w = P(2);
    const float* c1b1 = P(3);  const float* c1w1 = P(4);
    const float* c1b2 = P(5);  const float* c1w2 = P(6);
    const float* c2b1 = P(7);  const float* c2w1 = P(8);
    const float* c2b2 = P(9);  const float* c2w2 = P(10);
    const float* gb1  = P(11); const float* gw1  = P(12);
    const float* gb2  = P(13); const float* gw2  = P(14);
    const float* ib1  = P(15); const float* iw1  = P(16);
    const float* ib2  = P(17); const float* iw2  = P(18);
    const float* tb1  = P(19); const float* tw1  = P(20);
    const float* tb2  = P(21); const float* tw2  = P(22);
    const float* wb1  = P(23); const float* ww1  = P(24);
    const float* wb2  = P(25); const float* ww2  = P(26);

    rc_wall<<<B, 64, 0, stream>>>(wall_batch, ww1, wb1, ww2, wb2, wallf);
    rc_cond<<<N, 64, 0, stream>>>(category, geo, batch, cat_emb, cl_w, cl_b,
                                  gw1, gb1, gw2, gb2, wallf, cond, hnode);
    rc_init<<<N, 256, 0, stream>>>(x, actions, tars, iw1, ib1, iw2, ib2, hnode);

    // conv1
    rc_tw<<<(896 * 256 + 255) / 256, 256, 0, stream>>>(c1w1, W1t, 896);
    rc_tw<<<(256 * 256 + 255) / 256, 256, 0, stream>>>(c1w2, W2t, 256);
    rc_zero<<<(N * 256 + 255) / 256, 256, 0, stream>>>(out_enc, N * 256);
    rc_edgeconv<<<(E + ET - 1) / ET, 256, 0, stream>>>(hnode, W1t, c1b1, W2t, c1b2,
                                                       src, dst, out_enc, E);
    rc_post<<<N, 256, 0, stream>>>(out_enc, hnode);

    // conv2
    rc_tw<<<(896 * 256 + 255) / 256, 256, 0, stream>>>(c2w1, W1t, 896);
    rc_tw<<<(256 * 256 + 255) / 256, 256, 0, stream>>>(c2w2, W2t, 256);
    rc_zero<<<(N * 256 + 255) / 256, 256, 0, stream>>>(out_enc, N * 256);
    rc_edgeconv<<<(E + ET - 1) / ET, 256, 0, stream>>>(hnode, W1t, c2b1, W2t, c2b2,
                                                       src, dst, out_enc, E);

    rc_tail<<<N, 256, 0, stream>>>(out_enc, cond, tw1, tb1, tw2, tb2, q + (size_t)p * N);
  }
}
